// BrainInspiredAttention_86629490360396
// MI455X (gfx1250) — compile-verified, matrix operand order checked by reading, not run on hardware
//
#include <hip/hip_runtime.h>
#include <hip/hip_bf16.h>
#include <stdint.h>

typedef __bf16 bf16;
typedef __attribute__((ext_vector_type(16))) __bf16 v16bf;
typedef __attribute__((ext_vector_type(8)))  float  v8f;
typedef int v4i __attribute__((vector_size(16)));

union FragU { v16bf v; uint4 u[2]; };

#define DIMC   4096
#define NHEAD  4
#define HDIM   1024
#define NBATCH 8
#define SEQL   1024
#define ROWS   (NBATCH*SEQL)        /* 8192  */
#define QKVN   (3*DIMC)             /* 12288 */

#define BM 128
#define BN 128
#define BK 32
#define LDST 40                     /* padded LDS row stride in bf16 elems (80B, 16B-aligned) */

// ---------------------------------------------------------------------------
// Async global -> LDS copy (16 bytes/lane), ASYNCcnt-tracked on gfx1250.
// Falls back to a synchronous copy when the builtin is unavailable (host pass).
// ---------------------------------------------------------------------------
__device__ __forceinline__ void async_cp16(const bf16* g, bf16* l) {
#if __has_builtin(__builtin_amdgcn_global_load_async_to_lds_b128)
    __builtin_amdgcn_global_load_async_to_lds_b128(
        (__attribute__((address_space(1))) v4i*)g,
        (__attribute__((address_space(3))) v4i*)l, 0, 0);
#else
    *(uint4*)l = *(const uint4*)g;
#endif
}

__device__ __forceinline__ void async_wait0() {
#if __has_builtin(__builtin_amdgcn_s_wait_asynccnt)
    __builtin_amdgcn_s_wait_asynccnt(0);
#elif __has_builtin(__builtin_amdgcn_global_load_async_to_lds_b128)
    asm volatile("s_wait_asynccnt 0" ::: "memory");
#endif
}

// ---------------------------------------------------------------------------
// f32 -> bf16 conversion, 4 elements per thread
// ---------------------------------------------------------------------------
__global__ void __launch_bounds__(256) cvt_bf16_kernel(const float* __restrict__ in,
                                                       bf16* __restrict__ out, long n4) {
    long i = (long)blockIdx.x * blockDim.x + threadIdx.x;
    if (i < n4) {
        float4 f = ((const float4*)in)[i];
        union { bf16 h[4]; uint2 u; } p;
        p.h[0] = (bf16)f.x; p.h[1] = (bf16)f.y; p.h[2] = (bf16)f.z; p.h[3] = (bf16)f.w;
        ((uint2*)out)[i] = p.u;
    }
}

// ---------------------------------------------------------------------------
// Tiled WMMA GEMM:  C[row,col] = sum_k A[row,k] * B[col,k]  (+ bias / * colScale)
// A: [M,K] bf16 row-major (lda), B: [N,K] bf16 row-major (ldb)  -> C (ldc)
// Batched over blockIdx.z with (z/Hmod, z%Hmod) outer/inner strides.
// Block: 256 thr = 8 waves; block tile 128x128; wave tile 64x32 (4x2 WMMA tiles).
// LDS fills use global_load_async_to_lds_b128 with double buffering.
// ---------------------------------------------------------------------------
template<bool BIAS, bool SCALE, bool OUTF, bool OUTB>
__global__ void __launch_bounds__(256)
wmma_gemm_kernel(const bf16* __restrict__ A,  long lda, long aOuter, long aInner,
                 const bf16* __restrict__ Bm, long ldb, long bOuter, long bInner,
                 float* __restrict__ Cf, bf16* __restrict__ Cb,
                 long ldc, long cOuter, long cInner,
                 const float* __restrict__ bias,
                 const float* __restrict__ colScale, long csInner,
                 int K, int Hmod)
{
    __shared__ __align__(16) bf16 smem[2 * 2 * BM * LDST];   // [buf][A|B][BM][LDST] = 40 KB

    const int tid  = threadIdx.x;
    const int lane = tid & 31;
    const int wave = tid >> 5;
    const int idx  = lane & 15;       // row (A) / col (B) / col (C) within 16-tile
    const int half = lane >> 4;       // K-half selector (A/B), +8 row offset (C)
    const int waveM = wave >> 2;      // 0..1  -> 64-row slab
    const int waveN = wave & 3;       // 0..3  -> 32-col slab

    const int zo = blockIdx.z / Hmod, zi = blockIdx.z % Hmod;
    A  += (long)zo * aOuter + (long)zi * aInner;
    Bm += (long)zo * bOuter + (long)zi * bInner;
    const long cOff = (long)zo * cOuter + (long)zi * cInner;

    const int rowBase = blockIdx.y * BM;
    const int colBase = blockIdx.x * BN;

    // per-thread loader coordinates: 256 threads cover a 128x32 bf16 tile in
    // 512 16-byte chunks (2 per thread per matrix)
    const int lr = tid >> 2;          // rows 0..63 (second half at +64)
    const int lk = (tid & 3) * 8;     // k offset in elems (0,8,16,24)

    const bf16* aG0 = A  + (long)(rowBase + lr)      * lda + lk;
    const bf16* aG1 = A  + (long)(rowBase + 64 + lr) * lda + lk;
    const bf16* bG0 = Bm + (long)(colBase + lr)      * ldb + lk;
    const bf16* bG1 = Bm + (long)(colBase + 64 + lr) * ldb + lk;

    v8f acc[4][2] = {};
    const int nk = K / BK;

    // prologue: fill buffer 0
    {
        bf16* as = smem;
        bf16* bs = smem + BM * LDST;
        async_cp16(aG0, as + lr * LDST + lk);
        async_cp16(bG0, bs + lr * LDST + lk);
        async_cp16(aG1, as + (64 + lr) * LDST + lk);
        async_cp16(bG1, bs + (64 + lr) * LDST + lk);
    }
    async_wait0();
    __syncthreads();

    int buf = 0;
    for (int kb = 0; kb < nk; ++kb) {
        if (kb + 1 < nk) {                       // async-fill the other buffer
            int k1 = (kb + 1) * BK;
            bf16* as = smem + (buf ^ 1) * (2 * BM * LDST);
            bf16* bs = as + BM * LDST;
            async_cp16(aG0 + k1, as + lr * LDST + lk);
            async_cp16(bG0 + k1, bs + lr * LDST + lk);
            async_cp16(aG1 + k1, as + (64 + lr) * LDST + lk);
            async_cp16(bG1 + k1, bs + (64 + lr) * LDST + lk);
        }
        if (kb + 2 < nk) {                       // global_prefetch_b8, 2 steps ahead
            int k2 = (kb + 2) * BK;
            __builtin_prefetch(aG0 + k2, 0, 1);
            __builtin_prefetch(bG0 + k2, 0, 1);
        }

        const bf16* as = smem + buf * (2 * BM * LDST);
        const bf16* bs = as + BM * LDST;

        FragU aF[4], bF[2];
#pragma unroll
        for (int mt = 0; mt < 4; ++mt) {
            const bf16* r = as + (waveM * 64 + mt * 16 + idx) * LDST;
            aF[mt].u[0] = *(const uint4*)(r + 8 * half);        // K 0..7  (+8*half)
            aF[mt].u[1] = *(const uint4*)(r + 16 + 8 * half);   // K 16..23(+8*half)
        }
#pragma unroll
        for (int nt = 0; nt < 2; ++nt) {
            const bf16* r = bs + (waveN * 32 + nt * 16 + idx) * LDST;
            bF[nt].u[0] = *(const uint4*)(r + 8 * half);
            bF[nt].u[1] = *(const uint4*)(r + 16 + 8 * half);
        }

#pragma unroll
        for (int mt = 0; mt < 4; ++mt)
#pragma unroll
            for (int nt = 0; nt < 2; ++nt)
                acc[mt][nt] = __builtin_amdgcn_wmma_f32_16x16x32_bf16(
                    false, aF[mt].v, false, bF[nt].v,
                    (short)0, acc[mt][nt], false, false);

        if (kb + 1 < nk) async_wait0();          // drain DMA after compute overlap
        __syncthreads();
        buf ^= 1;
    }

    const float* cs = SCALE ? (colScale + (long)zi * csInner) : nullptr;
#pragma unroll
    for (int mt = 0; mt < 4; ++mt) {
#pragma unroll
        for (int nt = 0; nt < 2; ++nt) {
            int col = colBase + waveN * 32 + nt * 16 + idx;
            float s = SCALE ? cs[col] : 1.0f;
            float b = BIAS ? bias[col] : 0.0f;
#pragma unroll
            for (int r = 0; r < 8; ++r) {
                int row = rowBase + waveM * 64 + mt * 16 + r + 8 * half;
                float v = acc[mt][nt][r] * s + b;
                long a = cOff + (long)row * ldc + col;
                if (OUTF) Cf[a] = v;
                if (OUTB) Cb[a] = (bf16)v;
            }
        }
    }
}

// ---------------------------------------------------------------------------
// Split qkv f32 [8192,12288] into per-head bf16 q,k ([b,h,n,d]) and vT ([b,h,d,n])
// ---------------------------------------------------------------------------
__global__ void __launch_bounds__(256) split_qkv_kernel(const float* __restrict__ qkv,
                                                        bf16* __restrict__ q,
                                                        bf16* __restrict__ k,
                                                        bf16* __restrict__ vT) {
    long i = (long)blockIdx.x * blockDim.x + threadIdx.x;   // 0 .. B*H*N*D-1
    int d = (int)(i & (HDIM - 1));
    int n = (int)((i >> 10) & (SEQL - 1));
    int h = (int)((i >> 20) & (NHEAD - 1));
    int b = (int)(i >> 22);
    long rb = ((long)(b * SEQL + n)) * QKVN;
    float qf = qkv[rb + h * HDIM + d];
    float kf = qkv[rb + DIMC + h * HDIM + d];
    float vf = qkv[rb + 2 * DIMC + h * HDIM + d];
    long zh = (long)(b * NHEAD + h);
    q [zh * SEQL * HDIM + (long)n * HDIM + d] = (bf16)qf;
    k [zh * SEQL * HDIM + (long)n * HDIM + d] = (bf16)kf;
    vT[zh * HDIM * SEQL + (long)d * SEQL + n] = (bf16)vf;
}

// ---------------------------------------------------------------------------
// Row softmax over 1024 f32; writes attn f32 (d_out) and attn bf16 (ws)
// ---------------------------------------------------------------------------
__global__ void __launch_bounds__(256) softmax_kernel(const float* __restrict__ S,
                                                      float* __restrict__ attnF,
                                                      bf16* __restrict__ attnB) {
    __shared__ float red[8];
    long row = blockIdx.x;
    const float* s = S + row * SEQL;
    int tid = threadIdx.x;

    float x[4];
    float m = -1e30f;
#pragma unroll
    for (int j = 0; j < 4; ++j) { x[j] = s[tid + j * 256]; m = fmaxf(m, x[j]); }
#pragma unroll
    for (int o = 16; o > 0; o >>= 1) m = fmaxf(m, __shfl_xor(m, o, 32));
    if ((tid & 31) == 0) red[tid >> 5] = m;
    __syncthreads();
    float mm = red[0];
#pragma unroll
    for (int i = 1; i < 8; ++i) mm = fmaxf(mm, red[i]);

    float sum = 0.f;
#pragma unroll
    for (int j = 0; j < 4; ++j) { x[j] = __expf(x[j] - mm); sum += x[j]; }
#pragma unroll
    for (int o = 16; o > 0; o >>= 1) sum += __shfl_xor(sum, o, 32);
    __syncthreads();
    if ((tid & 31) == 0) red[tid >> 5] = sum;
    __syncthreads();
    float tot = 0.f;
#pragma unroll
    for (int i = 0; i < 8; ++i) tot += red[i];
    float inv = 1.0f / tot;

#pragma unroll
    for (int j = 0; j < 4; ++j) {
        float a = x[j] * inv;
        long o = row * SEQL + tid + j * 256;
        attnF[o] = a;
        attnB[o] = (bf16)a;
    }
}

// ---------------------------------------------------------------------------
extern "C" void kernel_launch(void* const* d_in, const int* in_sizes, int n_in,
                              void* d_out, int out_size, void* d_ws, size_t ws_size,
                              hipStream_t stream) {
    (void)in_sizes; (void)n_in; (void)out_size; (void)ws_size;

    const float* x     = (const float*)d_in[0];  // [8,1024,4096]
    const float* Wqkv  = (const float*)d_in[1];  // [12288,4096]
    const float* bqkv  = (const float*)d_in[2];  // [12288]
    const float* Wout  = (const float*)d_in[3];  // [4096,4096]
    const float* bout  = (const float*)d_in[4];  // [4096]
    const float* neuro = (const float*)d_in[5];  // [4,1024]

    float* out   = (float*)d_out;                      // [8192,4096]
    float* attnF = out + (long)ROWS * DIMC;            // [32,1024,1024]

    const long BH = (long)NBATCH * NHEAD;              // 32
    char* w = (char*)d_ws;
    bf16* xb    = (bf16*)w;  w += (long)ROWS * DIMC * 2;
    bf16* Wqkvb = (bf16*)w;  w += (long)QKVN * DIMC * 2;
    bf16* Woutb = (bf16*)w;  w += (long)DIMC * DIMC * 2;
    float* qkv  = (float*)w; w += (long)ROWS * QKVN * 4;   // 384 MiB
    float* scores = qkv;                                    // reuse (qkv dead after split)
    bf16* qb    = (bf16*)w;  w += BH * SEQL * HDIM * 2;
    bf16* kb    = (bf16*)w;  w += BH * SEQL * HDIM * 2;
    bf16* vTb   = (bf16*)w;  w += BH * HDIM * SEQL * 2;
    bf16* attnB = (bf16*)w;  w += BH * SEQL * SEQL * 2;
    bf16* ctxb  = (bf16*)w;  w += (long)ROWS * DIMC * 2;

    dim3 blk(256);

    // 1) f32 -> bf16 conversions
    cvt_bf16_kernel<<<dim3((ROWS * DIMC) / 1024), blk, 0, stream>>>(x, xb, (long)ROWS * DIMC / 4);
    cvt_bf16_kernel<<<dim3(((long)QKVN * DIMC) / 1024), blk, 0, stream>>>(Wqkv, Wqkvb, (long)QKVN * DIMC / 4);
    cvt_bf16_kernel<<<dim3(((long)DIMC * DIMC) / 1024), blk, 0, stream>>>(Wout, Woutb, (long)DIMC * DIMC / 4);

    // 2) qkv = x @ Wqkv^T + bqkv   (M=8192, N=12288, K=4096)
    wmma_gemm_kernel<true, false, true, false><<<dim3(QKVN / BN, ROWS / BM, 1), blk, 0, stream>>>(
        xb, DIMC, 0, 0, Wqkvb, DIMC, 0, 0,
        qkv, nullptr, QKVN, 0, 0, bqkv, nullptr, 0, DIMC, 1);

    // 3) split into heads (q,k row-major over d; v transposed)
    split_qkv_kernel<<<dim3((BH * SEQL * HDIM) / 256), blk, 0, stream>>>(qkv, qb, kb, vTb);

    // 4) scores = (q @ k^T) * neuro[h,col]   batched over 32 (b,h)
    wmma_gemm_kernel<false, true, true, false><<<dim3(SEQL / BN, SEQL / BM, BH), blk, 0, stream>>>(
        qb, HDIM, (long)NHEAD * SEQL * HDIM, (long)SEQL * HDIM,
        kb, HDIM, (long)NHEAD * SEQL * HDIM, (long)SEQL * HDIM,
        scores, nullptr, SEQL, (long)NHEAD * SEQL * SEQL, (long)SEQL * SEQL,
        nullptr, neuro, HDIM, HDIM, NHEAD);

    // 5) softmax rows; attn f32 -> d_out tail, attn bf16 -> ws
    softmax_kernel<<<dim3(BH * SEQL), blk, 0, stream>>>(scores, attnF, attnB);

    // 6) ctx = attn @ v  (B-operand = vT, row-major over m); write bf16 into [B,N,4096] layout
    wmma_gemm_kernel<false, false, false, true><<<dim3(HDIM / BN, SEQL / BM, BH), blk, 0, stream>>>(
        attnB, SEQL, (long)NHEAD * SEQL * SEQL, (long)SEQL * SEQL,
        vTb,   SEQL, (long)NHEAD * HDIM * SEQL, (long)HDIM * SEQL,
        nullptr, ctxb, DIMC, (long)SEQL * DIMC, (long)HDIM,
        nullptr, nullptr, 0, SEQL, NHEAD);

    // 7) out = ctx @ Wout^T + bout   (M=8192, N=4096, K=4096)
    wmma_gemm_kernel<true, false, true, false><<<dim3(DIMC / BN, ROWS / BM, 1), blk, 0, stream>>>(
        ctxb, DIMC, 0, 0, Woutb, DIMC, 0, 0,
        out, nullptr, DIMC, 0, 0, bout, nullptr, 0, DIMC, 1);
}
